// MultiHeadAttention_15298673508514
// MI455X (gfx1250) — matrix operand order repaired and checked by reading, not compiled, not run on hardware
//
#include <hip/hip_runtime.h>
#include <hip/hip_bf16.h>
#include <math.h>

#define D_MODEL_C 1024
#define NUM_HEADS_C 16
#define D_K_C 64
#define SEQ_C 2048
#define BATCH_C 2
#define ROWS_C (BATCH_C * SEQ_C)

typedef __bf16 bf16;
typedef __attribute__((ext_vector_type(16))) __bf16 v16bf;
typedef __attribute__((ext_vector_type(8)))  __bf16 bf16x8;
typedef __attribute__((ext_vector_type(4)))  __bf16 bf16x4;
typedef __attribute__((ext_vector_type(8)))  float   v8f;
typedef __attribute__((ext_vector_type(4)))  float   f32x4;
typedef __attribute__((ext_vector_type(4)))  unsigned v4u;
typedef __attribute__((ext_vector_type(8)))  int      v8i;
typedef __attribute__((ext_vector_type(4)))  int      v4i;

#define LOG2E 1.4426950408889634f

__device__ __forceinline__ bf16 f2bf(float f) { return static_cast<bf16>(f); }

__device__ __forceinline__ v8f wmma_bf16(v16bf a, v16bf b, v8f c) {
  return __builtin_amdgcn_wmma_f32_16x16x32_bf16(
      false, a, false, b, (short)0, c, false, false);
}

// A-fragment (16x32 bf16, M = lane%16): lanes<16 hold K = 0..7 & 16..23,
// lanes>=16 hold K = 8..15 & 24..31  ->  klo = 8*(lane>>4)
__device__ __forceinline__ v16bf frag_a(const bf16* rowk, int klo) {
  bf16x8 lo = *(const bf16x8*)(rowk + klo);
  bf16x8 hi = *(const bf16x8*)(rowk + 16 + klo);
  return __builtin_shufflevector(lo, hi, 0,1,2,3,4,5,6,7,8,9,10,11,12,13,14,15);
}
// B-fragment (32x16 bf16, N = lane%16): lanes<16 hold K = 0..7 & 16..23,
// lanes>=16 hold K = 8..15 & 24..31, read from a K-major (transposed) row.
__device__ __forceinline__ v16bf frag_b(const bf16* rowk, int hf) {
  bf16x8 lo = *(const bf16x8*)(rowk + hf * 8);
  bf16x8 hi = *(const bf16x8*)(rowk + 16 + hf * 8);
  return __builtin_shufflevector(lo, hi, 0,1,2,3,4,5,6,7,8,9,10,11,12,13,14,15);
}

// ---------------------------------------------------------------------------
// TDM: 2D bf16 tile (tile_d1 rows x tile_d0 elems) global -> LDS, with LDS
// row padding via D#.pad. Issue from one wave only; drain with s_wait_tensorcnt.
// ---------------------------------------------------------------------------
__device__ __forceinline__ void tdm_load_2d_bf16(
    unsigned lds_byte_addr, const void* gaddr,
    unsigned tensor_d0, unsigned tensor_d1,
    unsigned tile_d0, unsigned tile_d1, unsigned d0_stride,
    unsigned pad_interval_code, unsigned pad_amount_code)
{
  unsigned long long ga = (unsigned long long)gaddr;
  v4u g0;
  g0[0] = 1u;                                            // count=1, user mode
  g0[1] = lds_byte_addr;                                 // lds_addr
  g0[2] = (unsigned)(ga & 0xFFFFFFFFull);                // global_addr[31:0]
  g0[3] = (unsigned)((ga >> 32) & 0x1FFFFFFull) | (2u << 30); // addr[56:32] | type=2
  v8i g1;
  g1[0] = (int)((1u << 16) | (1u << 20) |                // data_size=2B, pad_enable
                (pad_interval_code << 22) | (pad_amount_code << 25));
  g1[1] = (int)((tensor_d0 & 0xFFFFu) << 16);            // tensor_dim0[15:0] @ 63:48
  g1[2] = (int)(((tensor_d0 >> 16) & 0xFFFFu) | ((tensor_d1 & 0xFFFFu) << 16));
  g1[3] = (int)(((tensor_d1 >> 16) & 0xFFFFu) | ((tile_d0 & 0xFFFFu) << 16));
  g1[4] = (int)(tile_d1 & 0xFFFFu);                      // tile_dim1 (tile_dim2=0)
  g1[5] = (int)d0_stride;                                // tensor_dim0_stride[31:0]
  g1[6] = 0;
  g1[7] = 0;
  v4i gz4 = {0, 0, 0, 0};
  v8i gz8 = {0, 0, 0, 0, 0, 0, 0, 0};
  __builtin_amdgcn_tensor_load_to_lds(g0, g1, gz4, gz4, gz8, 0);
}

__device__ __forceinline__ unsigned lds_addr_of(const void* p) {
  return (unsigned)(unsigned long long)p;  // low 32 bits of LDS aperture addr
}

// ---------------------------------------------------------------------------
// Kernel 1: Out[b,h,s,d] = (X @ W + bias), fp32 in, bf16 head-split out.
// Block tile 128x64, 8 waves (4x2), each wave 2x2 WMMA tiles, K-chunk 32.
// Global->register->LDS software pipeline hides HBM latency under WMMA.
// ---------------------------------------------------------------------------
#define BM 128
#define BN 64
#define BK 32
#define LDT (BK + 8)

__global__ __launch_bounds__(256) void proj_gemm_heads(
    const float* __restrict__ X, const float* __restrict__ W,
    const float* __restrict__ bias, bf16* __restrict__ Out)
{
  __shared__ bf16 As[BM][LDT];   // row-major X chunk (bf16)
  __shared__ bf16 Bs[BN][LDT];   // transposed W chunk: Bs[n][k]

  const int tid = threadIdx.x;
  const int lane = tid & 31;
  const int hf = lane >> 4, l16 = lane & 15;
  const int wid = tid >> 5;
  const int wm = wid & 3, wn = wid >> 2;
  const int m0 = blockIdx.y * BM, n0 = blockIdx.x * BN;

  v8f acc[2][2];
  #pragma unroll
  for (int tn = 0; tn < 2; ++tn) {
    float bv = bias[n0 + wn * 32 + tn * 16 + l16];
    #pragma unroll
    for (int tm = 0; tm < 2; ++tm)
      #pragma unroll
      for (int i = 0; i < 8; ++i) acc[tm][tn][i] = bv;
  }

  const int arow = tid >> 3, acg = (tid & 7) * 4;     // A: 4 rows x float4
  const int bk0  = tid >> 4, bn0 = (tid & 15) * 4;    // B: 2 rows x float4

  f32x4 xa[4], wb[2];
  #pragma unroll
  for (int j = 0; j < 4; ++j)
    xa[j] = *(const f32x4*)(X + (size_t)(m0 + arow + j * 32) * D_MODEL_C + acg);
  #pragma unroll
  for (int j = 0; j < 2; ++j)
    wb[j] = *(const f32x4*)(W + (size_t)(bk0 + j * 16) * D_MODEL_C + n0 + bn0);

  for (int k0 = 0; k0 < D_MODEL_C; k0 += BK) {
    // registers (chunk i) -> LDS
    #pragma unroll
    for (int j = 0; j < 4; ++j)
      *(bf16x4*)&As[arow + j * 32][acg] = __builtin_convertvector(xa[j], bf16x4);
    #pragma unroll
    for (int j = 0; j < 2; ++j)
      #pragma unroll
      for (int c = 0; c < 4; ++c) Bs[bn0 + c][bk0 + j * 16] = f2bf(wb[j][c]);
    __syncthreads();

    // issue global loads for chunk i+1: latency hidden under the WMMAs below
    if (k0 + BK < D_MODEL_C) {
      #pragma unroll
      for (int j = 0; j < 4; ++j)
        xa[j] = *(const f32x4*)(X + (size_t)(m0 + arow + j * 32) * D_MODEL_C +
                                k0 + BK + acg);
      #pragma unroll
      for (int j = 0; j < 2; ++j)
        wb[j] = *(const f32x4*)(W + (size_t)(k0 + BK + bk0 + j * 16) * D_MODEL_C +
                                n0 + bn0);
      if (k0 + 2 * BK < D_MODEL_C) {   // prefetch chunk i+2 toward L2/L0
        __builtin_prefetch(X + (size_t)(m0 + arow) * D_MODEL_C + k0 + 2 * BK + acg, 0, 3);
        __builtin_prefetch(W + (size_t)(k0 + 2 * BK + bk0) * D_MODEL_C + n0 + bn0, 0, 3);
      }
    }

    v16bf af[2], bfr[2];
    #pragma unroll
    for (int tm = 0; tm < 2; ++tm)
      af[tm] = frag_a(&As[wm * 32 + tm * 16 + l16][0], hf * 8);
    #pragma unroll
    for (int tn = 0; tn < 2; ++tn)
      bfr[tn] = frag_b(&Bs[wn * 32 + tn * 16 + l16][0], hf);
    #pragma unroll
    for (int tm = 0; tm < 2; ++tm)
      #pragma unroll
      for (int tn = 0; tn < 2; ++tn)
        acc[tm][tn] = wmma_bf16(af[tm], bfr[tn], acc[tm][tn]);
    __syncthreads();
  }

  #pragma unroll
  for (int tn = 0; tn < 2; ++tn) {
    int ng = n0 + wn * 32 + tn * 16 + l16;
    int h = ng >> 6, d = ng & 63;
    #pragma unroll
    for (int tm = 0; tm < 2; ++tm)
      #pragma unroll
      for (int i = 0; i < 8; ++i) {
        int r = m0 + wm * 32 + tm * 16 + i + hf * 8;
        int b = r >> 11, s = r & (SEQ_C - 1);
        Out[(((size_t)b * NUM_HEADS_C + h) * SEQ_C + s) * D_K_C + d] =
            f2bf(acc[tm][tn][i]);
      }
  }
}

// ---------------------------------------------------------------------------
// Kernel 2: flash attention per (b*H, 64-query block), 32-key inner blocks.
// K tiles: TDM ping-pong (issue kb+1 at loop top, s_wait_tensorcnt 1 in
// steady state, 0 on the last block). Scores live in LDS only.
// ---------------------------------------------------------------------------
#define QB 64
#define KB 32
#define NKB (SEQ_C / KB)

__global__ __launch_bounds__(256) void flash_attn(
    const bf16* __restrict__ Qm, const bf16* __restrict__ Km,
    const bf16* __restrict__ Vm, bf16* __restrict__ Ctx)
{
  __shared__ bf16  Qs[QB][D_K_C + 8];      // 64 x 72 (row = 128B + 16B pad)
  __shared__ bf16  Ks[2][KB][D_K_C + 8];   // double-buffered K tiles
  __shared__ bf16  Vts[D_K_C][KB + 8];     // transposed V: Vts[d][key]
  __shared__ float Ss[QB][KB + 4];         // fp32 scores
  __shared__ bf16  Ps[QB][KB + 8];         // softmaxed probs (bf16)
  __shared__ float corr[QB];
  __shared__ float lsum[QB];

  const int tid = threadIdx.x;
  const int lane = tid & 31;
  const int hf = lane >> 4, l16 = lane & 15;
  const int wid = tid >> 5;
  const int wm = wid & 3, wn = wid >> 2;   // 4 row-tiles x 2 key/dim-halves
  const int bh = blockIdx.y;               // b*H + h
  const int q0 = blockIdx.x * QB;

  const bf16* Qb = Qm + ((size_t)bh * SEQ_C + q0) * D_K_C;
  const bf16* Kb = Km + (size_t)bh * SEQ_C * D_K_C;
  const bf16* Vb = Vm + (size_t)bh * SEQ_C * D_K_C;

  // TDM: Q tile 64x64 + first K tile 32x64.  Row = 64 bf16 = 32 dwords
  // -> pad_interval code 4 (2^(4+1)=32 dw); pad 8 bf16 = 4 dw -> code 3.
  if (wid == 0) {
    tdm_load_2d_bf16(lds_addr_of(&Qs[0][0]), Qb, D_K_C, SEQ_C, D_K_C, QB,
                     D_K_C, 4u, 3u);
    tdm_load_2d_bf16(lds_addr_of(&Ks[0][0][0]), Kb, D_K_C, SEQ_C, D_K_C, KB,
                     D_K_C, 4u, 3u);
  }

  v8f oacc[2];
  #pragma unroll
  for (int t = 0; t < 2; ++t)
    #pragma unroll
    for (int i = 0; i < 8; ++i) oacc[t][i] = 0.f;
  float m_r = -INFINITY, l_r = 0.f;        // owned by threads tid < 64

  const int skey = tid >> 3, soff = (tid & 7) * 8;   // V staging coords

  for (int kb = 0; kb < NKB; ++kb) {
    const int idx = kb & 1;
    const bf16* Vt = Vb + (size_t)kb * KB * D_K_C;
    __builtin_prefetch(Vt + (size_t)skey * D_K_C + soff, 0, 3);

    if (wid == 0) {
      if (kb + 1 < NKB) {
        // ping-pong: K(kb+1) into the buffer last read in iteration kb-1
        tdm_load_2d_bf16(lds_addr_of(&Ks[idx ^ 1][0][0]),
                         Kb + (size_t)(kb + 1) * KB * D_K_C,
                         D_K_C, SEQ_C, D_K_C, KB, D_K_C, 4u, 3u);
        __builtin_amdgcn_s_wait_tensorcnt((short)1);  // K(kb) done, K(kb+1) in flight
      } else {
        __builtin_amdgcn_s_wait_tensorcnt((short)0);  // drain final tile
      }
    }
    __syncthreads();   // K(kb) (and Q on first iter) visible to all waves

    // S = Q @ K^T  (each wave: rows wm*16, keys wn*16), K-dim = 64 = 2 steps
    v8f sc;
    #pragma unroll
    for (int i = 0; i < 8; ++i) sc[i] = 0.f;
    #pragma unroll
    for (int kk = 0; kk < D_K_C; kk += 32) {
      v16bf aq = frag_a(&Qs[wm * 16 + l16][kk], hf * 8);
      v16bf bk = frag_b(&Ks[idx][wn * 16 + l16][kk], hf);
      sc = wmma_bf16(aq, bk, sc);
    }
    #pragma unroll
    for (int i = 0; i < 8; ++i)
      Ss[wm * 16 + i + hf * 8][wn * 16 + l16] = sc[i] * 0.125f; // 1/sqrt(64)
    __syncthreads();   // scores ready for softmax

    { // stage V transposed (all threads) ...
      bf16x8 vv = *(const bf16x8*)(Vt + (size_t)skey * D_K_C + soff);
      #pragma unroll
      for (int j = 0; j < 8; ++j) Vts[soff + j][skey] = vv[j];
    }
    if (tid < QB) { // ... while the first 2 waves run the online softmax
      int r = tid;
      float mx = m_r;
      #pragma unroll
      for (int k = 0; k < KB; ++k) mx = fmaxf(mx, Ss[r][k]);
      float c = exp2f((m_r - mx) * LOG2E);
      float s = 0.f;
      #pragma unroll
      for (int k = 0; k < KB; ++k) {
        float p = exp2f((Ss[r][k] - mx) * LOG2E);
        s += p;
        Ps[r][k] = f2bf(p);
      }
      m_r = mx;
      l_r = l_r * c + s;
      corr[r] = c;
    }
    __syncthreads();

    // O = O*corr + P @ V
    float cr[8];
    #pragma unroll
    for (int i = 0; i < 8; ++i) cr[i] = corr[wm * 16 + i + hf * 8];
    v16bf ap = frag_a(&Ps[wm * 16 + l16][0], hf * 8);
    #pragma unroll
    for (int tn = 0; tn < 2; ++tn) {
      v16bf bv = frag_b(&Vts[wn * 32 + tn * 16 + l16][0], hf);
      #pragma unroll
      for (int i = 0; i < 8; ++i) oacc[tn][i] *= cr[i];
      oacc[tn] = wmma_bf16(ap, bv, oacc[tn]);
    }
    __syncthreads();   // Ps/Vts consumed; next iteration may overwrite
  }

  if (tid < QB) lsum[tid] = l_r;
  __syncthreads();

  const int b = bh >> 4, h = bh & 15;
  #pragma unroll
  for (int tn = 0; tn < 2; ++tn) {
    int d = h * 64 + wn * 32 + tn * 16 + l16;
    #pragma unroll
    for (int i = 0; i < 8; ++i) {
      int rq = wm * 16 + i + hf * 8;
      float inv = 1.f / lsum[rq];
      int s = q0 + rq;
      Ctx[((size_t)b * SEQ_C + s) * D_MODEL_C + d] = f2bf(oacc[tn][i] * inv);
    }
  }
}

// ---------------------------------------------------------------------------
// Kernel 3: Out(fp32, row-major) = Ctx(bf16) @ Wo + bo.
// A chunks: TDM ping-pong into As[2]; W: register software pipeline.
// ---------------------------------------------------------------------------
__global__ __launch_bounds__(256) void out_proj_gemm(
    const bf16* __restrict__ X, const float* __restrict__ W,
    const float* __restrict__ bias, float* __restrict__ Out)
{
  __shared__ bf16 As[2][BM][LDT];
  __shared__ bf16 Bs[BN][LDT];

  const int tid = threadIdx.x;
  const int lane = tid & 31;
  const int hf = lane >> 4, l16 = lane & 15;
  const int wid = tid >> 5;
  const int wm = wid & 3, wn = wid >> 2;
  const int m0 = blockIdx.y * BM, n0 = blockIdx.x * BN;

  v8f acc[2][2];
  #pragma unroll
  for (int tn = 0; tn < 2; ++tn) {
    float bv = bias[n0 + wn * 32 + tn * 16 + l16];
    #pragma unroll
    for (int tm = 0; tm < 2; ++tm)
      #pragma unroll
      for (int i = 0; i < 8; ++i) acc[tm][tn][i] = bv;
  }

  const int bk0 = tid >> 4, bn0 = (tid & 15) * 4;

  // TDM first A chunk: tile 128 rows x 32 elems, row stride 1024 elems.
  // LDS row = 32 bf16 = 16 dwords -> pad_interval code 3; pad 4 dw -> code 3.
  if (wid == 0)
    tdm_load_2d_bf16(lds_addr_of(&As[0][0][0]),
                     X + (size_t)m0 * D_MODEL_C,
                     D_MODEL_C, ROWS_C, BK, BM, D_MODEL_C, 3u, 3u);

  f32x4 wb[2];
  #pragma unroll
  for (int j = 0; j < 2; ++j)
    wb[j] = *(const f32x4*)(W + (size_t)(bk0 + j * 16) * D_MODEL_C + n0 + bn0);

  for (int k0 = 0; k0 < D_MODEL_C; k0 += BK) {
    const int idx = (k0 / BK) & 1;
    // W registers (chunk i) -> LDS, transposed
    #pragma unroll
    for (int j = 0; j < 2; ++j)
      #pragma unroll
      for (int c = 0; c < 4; ++c) Bs[bn0 + c][bk0 + j * 16] = f2bf(wb[j][c]);

    if (wid == 0) {
      if (k0 + BK < D_MODEL_C) {
        tdm_load_2d_bf16(lds_addr_of(&As[idx ^ 1][0][0]),
                         X + (size_t)m0 * D_MODEL_C + k0 + BK,
                         D_MODEL_C, ROWS_C, BK, BM, D_MODEL_C, 3u, 3u);
        __builtin_amdgcn_s_wait_tensorcnt((short)1);  // chunk i done, i+1 in flight
      } else {
        __builtin_amdgcn_s_wait_tensorcnt((short)0);
      }
    }
    __syncthreads();

    // issue W loads for chunk i+1 under the WMMAs
    if (k0 + BK < D_MODEL_C) {
      #pragma unroll
      for (int j = 0; j < 2; ++j)
        wb[j] = *(const f32x4*)(W + (size_t)(k0 + BK + bk0 + j * 16) * D_MODEL_C +
                                n0 + bn0);
      if (k0 + 2 * BK < D_MODEL_C)
        __builtin_prefetch(W + (size_t)(k0 + 2 * BK + bk0) * D_MODEL_C + n0 + bn0, 0, 3);
    }

    v16bf af[2], bfr[2];
    #pragma unroll
    for (int tm = 0; tm < 2; ++tm)
      af[tm] = frag_a(&As[idx][wm * 32 + tm * 16 + l16][0], hf * 8);
    #pragma unroll
    for (int tn = 0; tn < 2; ++tn)
      bfr[tn] = frag_b(&Bs[wn * 32 + tn * 16 + l16][0], hf);
    #pragma unroll
    for (int tm = 0; tm < 2; ++tm)
      #pragma unroll
      for (int tn = 0; tn < 2; ++tn)
        acc[tm][tn] = wmma_bf16(af[tm], bfr[tn], acc[tm][tn]);
    __syncthreads();   // Bs consumed; As[idx] free for TDM in iteration i+1
  }

  #pragma unroll
  for (int tn = 0; tn < 2; ++tn) {
    int ng = n0 + wn * 32 + tn * 16 + l16;
    #pragma unroll
    for (int tm = 0; tm < 2; ++tm)
      #pragma unroll
      for (int i = 0; i < 8; ++i) {
        int r = m0 + wm * 32 + tm * 16 + i + hf * 8;
        Out[(size_t)r * D_MODEL_C + ng] = acc[tm][tn][i];
      }
  }
}

// ---------------------------------------------------------------------------
extern "C" void kernel_launch(void* const* d_in, const int* in_sizes, int n_in,
                              void* d_out, int out_size, void* d_ws, size_t ws_size,
                              hipStream_t stream) {
  (void)in_sizes; (void)n_in; (void)out_size; (void)ws_size;
  const float* q  = (const float*)d_in[0];
  const float* k  = (const float*)d_in[1];
  const float* v  = (const float*)d_in[2];
  const float* Wq = (const float*)d_in[3];
  const float* bq = (const float*)d_in[4];
  const float* Wk = (const float*)d_in[5];
  const float* bk = (const float*)d_in[6];
  const float* Wv = (const float*)d_in[7];
  const float* bv = (const float*)d_in[8];
  const float* Wo = (const float*)d_in[9];
  const float* bo = (const float*)d_in[10];
  float* out = (float*)d_out;

  char* ws = (char*)d_ws;
  const size_t tsz = (size_t)ROWS_C * D_MODEL_C * sizeof(bf16);  // 8 MiB each
  bf16* Qbf = (bf16*)(ws);
  bf16* Kbf = (bf16*)(ws + tsz);
  bf16* Vbf = (bf16*)(ws + 2 * tsz);
  bf16* Ctx = (bf16*)(ws + 3 * tsz);

  dim3 gproj(D_MODEL_C / BN, ROWS_C / BM);        // (16, 32)
  proj_gemm_heads<<<gproj, 256, 0, stream>>>(q, Wq, bq, Qbf);
  proj_gemm_heads<<<gproj, 256, 0, stream>>>(k, Wk, bk, Kbf);
  proj_gemm_heads<<<gproj, 256, 0, stream>>>(v, Wv, bv, Vbf);

  dim3 gattn(SEQ_C / QB, BATCH_C * NUM_HEADS_C);  // (32, 32)
  flash_attn<<<gattn, 256, 0, stream>>>(Qbf, Kbf, Vbf, Ctx);

  out_proj_gemm<<<gproj, 256, 0, stream>>>(Ctx, Wo, bo, out);
}
